// Attention_13795434955318
// MI455X (gfx1250) — compile-verified, matrix operand order checked by reading, not run on hardware
//
#include <hip/hip_runtime.h>
#include <hip/hip_bf16.h>

typedef __attribute__((ext_vector_type(16))) _Float16 v16h;
typedef __attribute__((ext_vector_type(8)))  float    v8f;
typedef __attribute__((ext_vector_type(4)))  unsigned int v4u;
typedef __attribute__((ext_vector_type(4)))  int      v4i;

#define B_  4
#define S_  1024
#define N_  1024
#define D_  1024
#define H_  16
#define HD_ 64
#define EPS_ 1e-5f

// ---- CDNA5 async global->LDS copies (ASYNCcnt path), guarded ------------
#if defined(__has_builtin)
#if __has_builtin(__builtin_amdgcn_global_load_async_to_lds_b128) && \
    __has_builtin(__builtin_amdgcn_s_wait_asynccnt)
#define HAVE_ASYNC 1
#endif
#endif
#ifndef HAVE_ASYNC
#define HAVE_ASYNC 0
#endif

#if HAVE_ASYNC
typedef __attribute__((address_space(1))) v4i* gp4_t;   // global <4 x i32>*
typedef __attribute__((address_space(3))) v4i* lp4_t;   // LDS    <4 x i32>*
#define ASYNC_WAIT(n) __builtin_amdgcn_s_wait_asynccnt(n)
#else
#define ASYNC_WAIT(n)
#endif

// 16-byte global -> LDS stage. Async path uses global_load_async_to_lds_b128
// (no VGPR round trip, tracked by ASYNCcnt; completes in issue order).
__device__ __forceinline__ void stage16(const _Float16* g, _Float16* l) {
#if HAVE_ASYNC
    gp4_t gp = (gp4_t)(unsigned long long)(uintptr_t)g;
    lp4_t lp = (lp4_t)(unsigned int)(uintptr_t)l;
    __builtin_amdgcn_global_load_async_to_lds_b128(gp, lp, 0, 0);
#else
    *(v4u*)l = *(const v4u*)g;
#endif
}

// -------------------------------------------------------------------------
// WMMA fragment load from a row-major f16 LDS tile.
//   lane L: row = base + (L&15), ksel = L>>4
//   v16h elems 0..7  = K[8*ksel   .. 8*ksel+7]
//   v16h elems 8..15 = K[16+8*ksel .. 16+8*ksel+7]
// -> two ds_load_b128 per fragment. Same pattern for A and B operands.
// -------------------------------------------------------------------------
__device__ inline v16h load_frag(const _Float16* rowp, int ksel) {
    union { v16h h; v4u u[2]; } r;
    r.u[0] = *(const v4u*)(rowp + 8 * ksel);
    r.u[1] = *(const v4u*)(rowp + 16 + 8 * ksel);
    return r.h;
}

__device__ inline v8f wmma_f16(v16h a, v16h b, v8f c) {
    return __builtin_amdgcn_wmma_f32_16x16x32_f16(false, a, false, b,
                                                  (short)0, c, false, false);
}

// -------------------------------------------------------------------------
// fp32 -> f16 bulk convert (once, so GEMMs stream f16 via the async path).
// -------------------------------------------------------------------------
__global__ __launch_bounds__(256) void f32_to_f16(const float* __restrict__ X,
                                                  _Float16* __restrict__ Y) {
    const size_t i = ((size_t)blockIdx.x * 256 + threadIdx.x) * 4;
    const float4 f = *(const float4*)(X + i);
    _Float16 tmp[4] = { (_Float16)f.x, (_Float16)f.y, (_Float16)f.z, (_Float16)f.w };
    *(unsigned long long*)(Y + i) = *(unsigned long long*)tmp;
}

// -------------------------------------------------------------------------
// GEMM: Y[M,E] = X[M,K] * W[E,K]^T, f16 in / fp32 accumulate.
// Block tile 128x128, K-step 64, double-buffered LDS: next tile's async
// loads stay in flight (wait asynccnt<=8) while WMMAs drain current tile.
// -------------------------------------------------------------------------
#define GST 72   // LDS row stride in halves (144B: 16B aligned, conflict-free)

__global__ __launch_bounds__(256) void gemm_f16(const _Float16* __restrict__ X,
                                                const _Float16* __restrict__ W,
                                                float* __restrict__ Y,
                                                int M, int K, int E) {
    __shared__ _Float16 Xs[2][128 * GST];
    __shared__ _Float16 Ws[2][128 * GST];
    const int tid = threadIdx.x, lane = tid & 31, w = tid >> 5;
    const int li = lane & 15, ksel = lane >> 4;
    const int wm = (w >> 1) * 32, wn = (w & 1) * 64;
    const int m0 = blockIdx.y * 128, e0 = blockIdx.x * 128;
    const int nk = K >> 6;

    v8f acc[2][4] = {};

    // stage 128x64 halves of X and W into buffer `buf`: 8 async b128/thread
    auto stage_tile = [&](int kt, int buf) {
#pragma unroll
        for (int c = 0; c < 4; c++) {
            const int id = tid + 256 * c;          // 0..1023
            const int row = id >> 3, cc = (id & 7) * 8;
            stage16(X + (size_t)(m0 + row) * K + kt * 64 + cc, &Xs[buf][row * GST + cc]);
            stage16(W + (size_t)(e0 + row) * K + kt * 64 + cc, &Ws[buf][row * GST + cc]);
        }
    };

    stage_tile(0, 0);
    if (nk > 1) stage_tile(1, 1);

    for (int kt = 0; kt < nk; kt++) {
        const int cur = kt & 1;
        // tile kt's 8 loads were issued before tile kt+1's; in-order
        // completion => waiting to <=8 leaves the prefetch in flight.
        if (kt + 1 < nk) ASYNC_WAIT(8); else ASYNC_WAIT(0);
        __syncthreads();

        v16h a[2][2], bf[4][2];
#pragma unroll
        for (int i = 0; i < 2; i++)
#pragma unroll
            for (int kk = 0; kk < 2; kk++)
                a[i][kk] = load_frag(&Xs[cur][(wm + i * 16 + li) * GST + kk * 32], ksel);
#pragma unroll
        for (int j = 0; j < 4; j++)
#pragma unroll
            for (int kk = 0; kk < 2; kk++)
                bf[j][kk] = load_frag(&Ws[cur][(wn + j * 16 + li) * GST + kk * 32], ksel);

#pragma unroll
        for (int kk = 0; kk < 2; kk++)
#pragma unroll
            for (int i = 0; i < 2; i++)
#pragma unroll
                for (int j = 0; j < 4; j++)
                    acc[i][j] = wmma_f16(a[i][kk], bf[j][kk], acc[i][j]);
        __syncthreads();                 // all waves done reading buf `cur`
        if (kt + 2 < nk) stage_tile(kt + 2, cur);
    }

    // C/D layout: VGPR r -> row (r + 8*ksel), lane li -> column
#pragma unroll
    for (int i = 0; i < 2; i++)
#pragma unroll
        for (int j = 0; j < 4; j++)
#pragma unroll
            for (int r = 0; r < 8; r++) {
                const int m = m0 + wm + i * 16 + r + 8 * ksel;
                const int e = e0 + wn + j * 16 + li;
                Y[(size_t)m * E + e] = acc[i][j][r];
            }
}

// -------------------------------------------------------------------------
// Per-head LayerNorm of q (64-elem rows). Reference's "no-transpose reshape"
// is flat reinterpretation: rows stay contiguous 64-float chunks, identity
// row mapping. scale = HD^-0.5 folded in. One wave per row.
// -------------------------------------------------------------------------
__global__ __launch_bounds__(256) void ln_head_q(const float* __restrict__ X,
                                                 const float* __restrict__ g,
                                                 const float* __restrict__ be,
                                                 _Float16* __restrict__ Y,
                                                 float scale) {
    const int w = threadIdx.x >> 5, lane = threadIdx.x & 31;
    const size_t row = (size_t)blockIdx.x * 8 + w;
    const float2 f = *(const float2*)(X + row * 64 + lane * 2);
    float s = f.x + f.y;
#pragma unroll
    for (int off = 16; off >= 1; off >>= 1) s += __shfl_xor(s, off, 32);
    const float mean = s * (1.0f / 64.0f);
    const float d0 = f.x - mean, d1 = f.y - mean;
    float q = d0 * d0 + d1 * d1;
#pragma unroll
    for (int off = 16; off >= 1; off >>= 1) q += __shfl_xor(q, off, 32);
    const float r = rsqrtf(q * (1.0f / 64.0f) + EPS_);
    Y[row * 64 + lane * 2 + 0] = (_Float16)((d0 * r * g[lane * 2 + 0] + be[lane * 2 + 0]) * scale);
    Y[row * 64 + lane * 2 + 1] = (_Float16)((d1 * r * g[lane * 2 + 1] + be[lane * 2 + 1]) * scale);
}

// -------------------------------------------------------------------------
// k/v path: gather (b,n,h) rows -> head-major (b,h,n) layout, optional LN.
// -------------------------------------------------------------------------
__global__ __launch_bounds__(256) void permute_head_kv(const float* __restrict__ X,
                                                       const float* __restrict__ g,
                                                       const float* __restrict__ be,
                                                       _Float16* __restrict__ Y,
                                                       int doLN) {
    const int w = threadIdx.x >> 5, lane = threadIdx.x & 31;
    const size_t rid = (size_t)blockIdx.x * 8 + w;   // output-major (b,h,n)
    const int b = (int)(rid / (H_ * N_));
    const int h = (int)((rid / N_) % H_);
    const int n = (int)(rid % N_);
    const float2 f = *(const float2*)(X + ((size_t)(b * N_ + n) * D_ + h * HD_) + lane * 2);
    float y0, y1;
    if (doLN) {
        float s = f.x + f.y;
#pragma unroll
        for (int off = 16; off >= 1; off >>= 1) s += __shfl_xor(s, off, 32);
        const float mean = s * (1.0f / 64.0f);
        const float d0 = f.x - mean, d1 = f.y - mean;
        float q = d0 * d0 + d1 * d1;
#pragma unroll
        for (int off = 16; off >= 1; off >>= 1) q += __shfl_xor(q, off, 32);
        const float r = rsqrtf(q * (1.0f / 64.0f) + EPS_);
        y0 = d0 * r * g[lane * 2 + 0] + be[lane * 2 + 0];
        y1 = d1 * r * g[lane * 2 + 1] + be[lane * 2 + 1];
    } else {
        y0 = f.x; y1 = f.y;
    }
    Y[rid * 64 + lane * 2 + 0] = (_Float16)y0;
    Y[rid * 64 + lane * 2 + 1] = (_Float16)y1;
}

// -------------------------------------------------------------------------
// Flash attention per (b,h). Block = 8 waves; wave owns 16 query rows.
// N chunked by 64. K double-buffered via async copies (next chunk in flight
// during compute); V transposed through VGPRs. Online softmax in C-layout,
// P converted C->A layout through a per-wave LDS buffer.
// -------------------------------------------------------------------------
#define AST 72

__global__ __launch_bounds__(256) void flash_attn(const _Float16* __restrict__ qn,
                                                  const _Float16* __restrict__ kn,
                                                  const _Float16* __restrict__ vn,
                                                  float* __restrict__ O) {
    __shared__ _Float16 Kb[2][64 * AST];
    __shared__ _Float16 Vt[64 * AST];
    __shared__ _Float16 Pb[8 * 16 * AST];

    const int tid = threadIdx.x, lane = tid & 31, w = tid >> 5;
    const int li = lane & 15, ksel = lane >> 4;
    const int bh = blockIdx.y;                 // b*H + h
    const int b = bh >> 4, h = bh & 15;
    const int s0 = blockIdx.x * 128 + w * 16;  // wave's query-row base

    v16h qa[2];
    {
        const _Float16* p = qn + ((size_t)bh * S_ + s0 + li) * HD_;
        qa[0] = load_frag(p, ksel);
        qa[1] = load_frag(p + 32, ksel);
    }

    v8f o[4] = {};
    float mrow[8], lrow[8];
#pragma unroll
    for (int r = 0; r < 8; r++) { mrow[r] = -1e30f; lrow[r] = 0.0f; }

    const _Float16* kbase = kn + (size_t)bh * N_ * HD_;
    const _Float16* vbase = vn + (size_t)bh * N_ * HD_;
    const int row = tid >> 2, cs = (tid & 3) * 16;
    const int nch = N_ / 64;                   // 16

    // prologue: async-stage K chunk 0
    {
        const _Float16* ksrc = kbase + (size_t)row * HD_ + cs;
        stage16(ksrc,     &Kb[0][row * AST + cs]);
        stage16(ksrc + 8, &Kb[0][row * AST + cs + 8]);
    }

    for (int jx = 0; jx < nch; jx++) {
        const int cur = jx & 1;
        // prefetch next K chunk into the other buffer (stays in flight)
        if (jx + 1 < nch) {
            const _Float16* ksrc = kbase + (size_t)((jx + 1) * 64 + row) * HD_ + cs;
            stage16(ksrc,     &Kb[cur ^ 1][row * AST + cs]);
            stage16(ksrc + 8, &Kb[cur ^ 1][row * AST + cs + 8]);
        }
        // stage V chunk transposed via VGPRs (needs data in registers anyway)
        {
            union { v4u u[2]; _Float16 hh[16]; } d;
            const v4u* vs = (const v4u*)(vbase + (size_t)(jx * 64 + row) * HD_ + cs);
            d.u[0] = vs[0]; d.u[1] = vs[1];
#pragma unroll
            for (int i = 0; i < 16; i++)
                Vt[(cs + i) * AST + row] = d.hh[i];
        }
        // chunk jx's 2 loads precede chunk jx+1's 2 -> wait to <=2
        if (jx + 1 < nch) ASYNC_WAIT(2); else ASYNC_WAIT(0);
        __syncthreads();

        // ---- scores S = q * K^T for 4 key subtiles ----
        v8f sacc[4];
#pragma unroll
        for (int t = 0; t < 4; t++) {
            const _Float16* kp = &Kb[cur][(t * 16 + li) * AST];
            v8f z = {};
            z = wmma_f16(qa[0], load_frag(kp, ksel), z);
            sacc[t] = wmma_f16(qa[1], load_frag(kp + 32, ksel), z);
        }

        // ---- online softmax (rows r+8*ksel, width-16 reductions) ----
        float pval[4][8];
#pragma unroll
        for (int r = 0; r < 8; r++) {
            float vm = fmaxf(fmaxf(sacc[0][r], sacc[1][r]),
                             fmaxf(sacc[2][r], sacc[3][r]));
#pragma unroll
            for (int off = 8; off >= 1; off >>= 1) vm = fmaxf(vm, __shfl_xor(vm, off, 16));
            const float mnew = fmaxf(mrow[r], vm);
            const float corr = __expf(mrow[r] - mnew);
            mrow[r] = mnew;
            float rsum = 0.0f;
#pragma unroll
            for (int t = 0; t < 4; t++) {
                const float p = __expf(sacc[t][r] - mnew);
                pval[t][r] = p;
                rsum += p;
            }
#pragma unroll
            for (int off = 8; off >= 1; off >>= 1) rsum += __shfl_xor(rsum, off, 16);
            lrow[r] = lrow[r] * corr + rsum;
#pragma unroll
            for (int u = 0; u < 4; u++) o[u][r] *= corr;
        }

        // ---- P: C-layout -> per-wave LDS -> A-layout fragments ----
        _Float16* pb = &Pb[w * 16 * AST];
#pragma unroll
        for (int t = 0; t < 4; t++)
#pragma unroll
            for (int r = 0; r < 8; r++)
                pb[(r + 8 * ksel) * AST + t * 16 + li] = (_Float16)pval[t][r];

        v16h pa[2];
        {
            const _Float16* pp = &pb[li * AST];
            pa[0] = load_frag(pp, ksel);
            pa[1] = load_frag(pp + 32, ksel);
        }

        // ---- O += P * V  (V^T staged: B-fragment rows are hd, cols keys) ----
#pragma unroll
        for (int u = 0; u < 4; u++) {
            const _Float16* vp = &Vt[(u * 16 + li) * AST];
            o[u] = wmma_f16(pa[0], load_frag(vp, ksel), o[u]);
            o[u] = wmma_f16(pa[1], load_frag(vp + 32, ksel), o[u]);
        }
        __syncthreads();                 // all waves done with Vt / Kb[cur]
    }

    // ---- epilogue: normalize, write (B,S,D) fp32 ----
#pragma unroll
    for (int u = 0; u < 4; u++)
#pragma unroll
        for (int r = 0; r < 8; r++) {
            const int srow = s0 + r + 8 * ksel;
            const int dcol = h * HD_ + u * 16 + li;
            O[((size_t)b * S_ + srow) * D_ + dcol] = o[u][r] / lrow[r];
        }
}

// -------------------------------------------------------------------------
// Output LayerNorm over D=1024: one block per row, f16 output for final GEMM.
// -------------------------------------------------------------------------
__global__ __launch_bounds__(256) void out_ln(const float* __restrict__ X,
                                              const float* __restrict__ g,
                                              const float* __restrict__ be,
                                              _Float16* __restrict__ Y) {
    __shared__ float rs[8], rs2[8];
    const int tid = threadIdx.x, lane = tid & 31, w = tid >> 5;
    const size_t row = blockIdx.x;
    const float4 f = *(const float4*)(X + row * D_ + tid * 4);
    float s = f.x + f.y + f.z + f.w;
    float s2 = f.x * f.x + f.y * f.y + f.z * f.z + f.w * f.w;
#pragma unroll
    for (int off = 16; off >= 1; off >>= 1) {
        s += __shfl_xor(s, off, 32);
        s2 += __shfl_xor(s2, off, 32);
    }
    if (lane == 0) { rs[w] = s; rs2[w] = s2; }
    __syncthreads();
    float ts = 0.0f, ts2 = 0.0f;
#pragma unroll
    for (int i = 0; i < 8; i++) { ts += rs[i]; ts2 += rs2[i]; }
    const float mean = ts * (1.0f / D_);
    const float var = ts2 * (1.0f / D_) - mean * mean;
    const float r = rsqrtf(var + EPS_);
    const float xv[4] = { f.x, f.y, f.z, f.w };
    _Float16 tmp[4];
#pragma unroll
    for (int i = 0; i < 4; i++) {
        const int c = tid * 4 + i;
        tmp[i] = (_Float16)((xv[i] - mean) * r * g[c] + be[c]);
    }
    *(unsigned long long*)(Y + row * D_ + tid * 4) = *(unsigned long long*)tmp;
}

// -------------------------------------------------------------------------
extern "C" void kernel_launch(void* const* d_in, const int* in_sizes, int n_in,
                              void* d_out, int out_size, void* d_ws, size_t ws_size,
                              hipStream_t stream) {
    const float* x_q  = (const float*)d_in[0];
    const float* x_k  = (const float*)d_in[1];
    const float* x_v  = (const float*)d_in[2];
    const float* Wq   = (const float*)d_in[3];
    const float* Wk   = (const float*)d_in[4];
    const float* Wv   = (const float*)d_in[5];
    const float* Wp   = (const float*)d_in[6];
    const float* qn_g = (const float*)d_in[7];
    const float* qn_b = (const float*)d_in[8];
    const float* kn_g = (const float*)d_in[9];
    const float* kn_b = (const float*)d_in[10];
    const float* on_g = (const float*)d_in[11];
    const float* on_b = (const float*)d_in[12];

    const size_t MB = 1024ull * 1024ull;
    char* ws = (char*)d_ws;
    float*    qlin = (float*)(ws + 0 * MB);       // 16MB, reused as O
    float*    klin = (float*)(ws + 16 * MB);      // 16MB, reused as xln
    float*    vlin = (float*)(ws + 32 * MB);      // 16MB
    _Float16* qh   = (_Float16*)(ws + 48 * MB);   // 8MB
    _Float16* kh   = (_Float16*)(ws + 56 * MB);   // 8MB
    _Float16* vh   = (_Float16*)(ws + 64 * MB);   // 8MB
    _Float16* xqh  = (_Float16*)(ws + 72 * MB);   // 8MB
    _Float16* xkh  = (_Float16*)(ws + 80 * MB);   // 8MB
    _Float16* xvh  = (_Float16*)(ws + 88 * MB);   // 8MB
    _Float16* Wqh  = (_Float16*)(ws + 96 * MB);   // 2MB
    _Float16* Wkh  = (_Float16*)(ws + 98 * MB);   // 2MB
    _Float16* Wvh  = (_Float16*)(ws + 100 * MB);  // 2MB
    _Float16* Wph  = (_Float16*)(ws + 102 * MB);  // 2MB
    float*    Obuf = qlin;
    _Float16* xln  = (_Float16*)klin;

    const int M  = B_ * S_;                       // 4096
    const int NX = M * D_;                        // 4M elems
    const int NW = D_ * D_;                       // 1M elems

    // one-time fp32 -> f16 conversion of GEMM operands
    f32_to_f16<<<NX / 1024, 256, 0, stream>>>(x_q, xqh);
    f32_to_f16<<<NX / 1024, 256, 0, stream>>>(x_k, xkh);
    f32_to_f16<<<NX / 1024, 256, 0, stream>>>(x_v, xvh);
    f32_to_f16<<<NW / 1024, 256, 0, stream>>>(Wq, Wqh);
    f32_to_f16<<<NW / 1024, 256, 0, stream>>>(Wk, Wkh);
    f32_to_f16<<<NW / 1024, 256, 0, stream>>>(Wv, Wvh);
    f32_to_f16<<<NW / 1024, 256, 0, stream>>>(Wp, Wph);

    const dim3 gg(D_ / 128, M / 128);             // (8, 32)
    gemm_f16<<<gg, 256, 0, stream>>>(xqh, Wqh, qlin, M, D_, D_);
    gemm_f16<<<gg, 256, 0, stream>>>(xkh, Wkh, klin, M, D_, D_);
    gemm_f16<<<gg, 256, 0, stream>>>(xvh, Wvh, vlin, M, D_, D_);

    const int rows64 = B_ * H_ * S_;              // 65536
    ln_head_q<<<rows64 / 8, 256, 0, stream>>>(qlin, qn_g, qn_b, qh, 0.125f);
    permute_head_kv<<<rows64 / 8, 256, 0, stream>>>(klin, kn_g, kn_b, kh, 1);
    permute_head_kv<<<rows64 / 8, 256, 0, stream>>>(vlin, kn_g, kn_b, vh, 0);

    flash_attn<<<dim3(S_ / 128, B_ * H_), 256, 0, stream>>>(qh, kh, vh, Obuf);

    out_ln<<<M, 256, 0, stream>>>(Obuf, on_g, on_b, xln);

    gemm_f16<<<gg, 256, 0, stream>>>(xln, Wph, (float*)d_out, M, D_, D_);
}
